// CausalSelfAttention_68796786147934
// MI455X (gfx1250) — matrix operand order repaired and checked by reading, not compiled, not run on hardware
//
#include <hip/hip_runtime.h>
#include <hip/hip_bf16.h>
#include <math.h>

// ---------------------------------------------------------------------------
// Types matching the CDNA5 WMMA builtin signatures (wave32).
// ---------------------------------------------------------------------------
typedef _Float16 half_t;
typedef __attribute__((ext_vector_type(16))) _Float16 v16h;
typedef __attribute__((ext_vector_type(8)))  float    v8f;
typedef __attribute__((ext_vector_type(4)))  _Float16 h4;
typedef __attribute__((ext_vector_type(4)))  int      v4i;

#define HEAD_DIM   128
#define N_HEADS    16
#define N_KV_HEADS 4
#define B_SZ       4
#define T_SZ       2048
#define D_SZ       2048
#define BT_SZ      (B_SZ * T_SZ)            // 8192
#define KV_D       (N_KV_HEADS * HEAD_DIM)  // 512

// ---------------------------------------------------------------------------
// CDNA5 async global->LDS copies (ASYNCcnt path, no VGPR data round-trip).
// Builtin signature (per hipcc diagnostic): (int4 AS1*, int4 AS3*, imm, imm).
// Guarded so we fall back to the synchronous staging path if the toolchain
// lacks the builtins.
// ---------------------------------------------------------------------------
#if defined(__has_builtin)
#if __has_builtin(__builtin_amdgcn_global_load_async_to_lds_b128) && \
    __has_builtin(__builtin_amdgcn_s_wait_asynccnt)
#define USE_ASYNC_LDS 1
#endif
#endif
#ifndef USE_ASYNC_LDS
#define USE_ASYNC_LDS 0
#endif

#define AS1 __attribute__((address_space(1)))
#define AS3 __attribute__((address_space(3)))

__device__ __forceinline__ void async_copy_b128(half_t* lds_dst, const half_t* gsrc) {
#if USE_ASYNC_LDS
  __builtin_amdgcn_global_load_async_to_lds_b128((AS1 v4i*)gsrc, (AS3 v4i*)lds_dst, 0, 0);
#else
  *(uint4*)lds_dst = *(const uint4*)gsrc;
#endif
}

template <int N>
__device__ __forceinline__ void async_wait() {
#if USE_ASYNC_LDS
  __builtin_amdgcn_s_wait_asynccnt(N);
#endif
}

// ---------------------------------------------------------------------------
// Fragment gathers following the CDNA5 ISA VGPR layouts (CDNA5 ISA §7.12.2).
// A 16x32 f16: lanes 0-15 row M=lane, halves 0..7 = K 0..7, 8..15 = K 16..23;
//              lanes 16-31 same rows, K offset +8.
// B 32x16 f16: lane holds col N=lane&15; same K layout as A (0..7,16..23 / +8).
// C/D 16x16 f32: VGPR v, lanes 0-15 -> M=v, lanes 16-31 -> M=v+8; N=lane&15.
// ---------------------------------------------------------------------------
__device__ __forceinline__ v16h frag_a(const half_t* p, int ld, int mbase, int kbase) {
  const int lane = threadIdx.x & 31;
  const half_t* s = p + (size_t)(mbase + (lane & 15)) * ld + kbase + ((lane >> 4) << 3);
  v16h a;
#pragma unroll
  for (int i = 0; i < 8; ++i) { a[i] = s[i]; a[8 + i] = s[16 + i]; }
  return a;
}

__device__ __forceinline__ v16h frag_b(const half_t* p, int ld, int nbase, int kbase) {
  const int lane = threadIdx.x & 31;
  const half_t* s = p + (size_t)(nbase + (lane & 15)) * ld + kbase + ((lane >> 4) << 3);
  v16h b;
#pragma unroll
  for (int i = 0; i < 8; ++i) { b[i] = s[i]; b[8 + i] = s[16 + i]; }
  return b;
}

__device__ __forceinline__ v8f wmma_f16(v16h a, v16h b, v8f c) {
  // (neg_a, A, neg_b, B, c_mod, C, reuse_a, reuse_b)
  return __builtin_amdgcn_wmma_f32_16x16x32_f16(false, a, false, b, (short)0, c, false, false);
}

// ---------------------------------------------------------------------------
// Elementwise f32 -> f16 conversion (vectorized 4-wide).
// ---------------------------------------------------------------------------
__global__ __launch_bounds__(256) void f32_to_f16_kernel(const float4* __restrict__ s,
                                                         h4* __restrict__ d, long n4) {
  long i = (long)blockIdx.x * blockDim.x + threadIdx.x;
  if (i >= n4) return;
  float4 f = s[i];
  h4 o;
  o.x = (_Float16)f.x; o.y = (_Float16)f.y; o.z = (_Float16)f.z; o.w = (_Float16)f.w;
  d[i] = o;
}

// ---------------------------------------------------------------------------
// GEMM: C[M,N] = A[M,K] * B[N,K]^T  (both operands K-contiguous, f16 in,
// f32 accumulate via v_wmma_f32_16x16x32_f16). 128x128 tile per 256-thread
// block (8 waves in 4x2), K-step 32. Double-buffered LDS filled with async
// global->LDS DMA: tile k+1 streams in while tile k is computed.
// ---------------------------------------------------------------------------
template <typename OutT>
__global__ __launch_bounds__(256) void gemm_nt_kernel(const half_t* __restrict__ A,
                                                      const half_t* __restrict__ Bw,
                                                      OutT* __restrict__ C,
                                                      int M, int N, int K) {
  __shared__ __align__(16) half_t As[2][128 * 40];
  __shared__ __align__(16) half_t Bs[2][128 * 40];
  const int tid  = threadIdx.x;
  const int wid  = tid >> 5;
  const int lane = tid & 31;
  const int m0 = blockIdx.y * 128;
  const int n0 = blockIdx.x * 128;
  const int wm = (wid & 3) * 32;   // wave M offset in tile
  const int wn = (wid >> 2) * 64;  // wave N offset in tile

  v8f zero = {};
  v8f acc[2][4];
#pragma unroll
  for (int mi = 0; mi < 2; ++mi)
#pragma unroll
    for (int ni = 0; ni < 4; ++ni) acc[mi][ni] = zero;

  const int row = tid >> 1;          // 0..127
  const int hs  = (tid & 1) << 4;    // 0 or 16 halves
  const half_t* gA = A  + (size_t)(m0 + row) * K + hs;
  const half_t* gB = Bw + (size_t)(n0 + row) * K + hs;
  const int lo = row * 40 + hs;

  // Issue one tile's worth of async copies (4 x b128 per thread).
  auto issue = [&](int k0, int buf) {
    async_copy_b128(&As[buf][lo],     gA + k0);
    async_copy_b128(&As[buf][lo + 8], gA + k0 + 8);
    async_copy_b128(&Bs[buf][lo],     gB + k0);
    async_copy_b128(&Bs[buf][lo + 8], gB + k0 + 8);
  };

  const int nk = K / 32;
  issue(0, 0);
  for (int ki = 0; ki < nk; ++ki) {
    const int buf = ki & 1;
    if (ki + 1 < nk) {
      issue((ki + 1) * 32, buf ^ 1);
      async_wait<4>();   // oldest tile (ki) landed; next tile still in flight
    } else {
      async_wait<0>();
    }
    __syncthreads();

    v16h af[2], bf[4];
#pragma unroll
    for (int mi = 0; mi < 2; ++mi) af[mi] = frag_a(As[buf], 40, wm + mi * 16, 0);
#pragma unroll
    for (int ni = 0; ni < 4; ++ni) bf[ni] = frag_b(Bs[buf], 40, wn + ni * 16, 0);
#pragma unroll
    for (int mi = 0; mi < 2; ++mi)
#pragma unroll
      for (int ni = 0; ni < 4; ++ni) acc[mi][ni] = wmma_f16(af[mi], bf[ni], acc[mi][ni]);
    __syncthreads();   // protect buf from being refilled before all waves read it
  }

#pragma unroll
  for (int mi = 0; mi < 2; ++mi)
#pragma unroll
    for (int ni = 0; ni < 4; ++ni) {
      const int r = m0 + wm + mi * 16 + ((lane >> 4) << 3);
      const int c = n0 + wn + ni * 16 + (lane & 15);
#pragma unroll
      for (int v = 0; v < 8; ++v)
        C[(size_t)(r + v) * N + c] = (OutT)acc[mi][ni][v];
    }
}

// ---------------------------------------------------------------------------
// Fused per-head RMSNorm + RoPE (+ optional per-head gain). One wave per
// (b,t,head); lane handles rotary pairs i=lane and i=lane+32.
// log2_base already includes the >train-length extrapolation rescale.
// ---------------------------------------------------------------------------
__global__ __launch_bounds__(256) void rmsnorm_rope_kernel(const float* __restrict__ src,
                                                           half_t* __restrict__ dst,
                                                           const float* __restrict__ gain,
                                                           int H, int T, float log2_base) {
  const int wid  = threadIdx.x >> 5;
  const int lane = threadIdx.x & 31;
  const long idx = (long)blockIdx.x * 8 + wid;   // over B*T*H
  const int  h   = (int)(idx % H);
  const long bt  = idx / H;
  const int  t   = (int)(bt % T);
  const float* x = src + idx * HEAD_DIM;
  half_t*      y = dst + idx * HEAD_DIM;

  const float x1a = x[lane],      x1b = x[lane + 32];
  const float x2a = x[lane + 64], x2b = x[lane + 96];
  float ss = x1a * x1a + x1b * x1b + x2a * x2a + x2b * x2b;
#pragma unroll
  for (int off = 16; off >= 1; off >>= 1) ss += __shfl_xor(ss, off, 32);
  const float r = rsqrtf(ss * (1.0f / 128.0f) + 1.1920929e-7f);
  const float g = r * (gain ? gain[h] : 1.0f);

  {
    const float inv = exp2f(-((float)(2 * lane) * (1.0f / 128.0f)) * log2_base);
    const float ang = (float)t * inv;
    float sn, cs;
    __sincosf(ang, &sn, &cs);
    y[lane]      = (half_t)(( x1a * cs + x2a * sn) * g);
    y[lane + 64] = (half_t)((-x1a * sn + x2a * cs) * g);
  }
  {
    const int i = lane + 32;
    const float inv = exp2f(-((float)(2 * i) * (1.0f / 128.0f)) * log2_base);
    const float ang = (float)t * inv;
    float sn, cs;
    __sincosf(ang, &sn, &cs);
    y[i]      = (half_t)(( x1b * cs + x2b * sn) * g);
    y[i + 64] = (half_t)((-x1b * sn + x2b * cs) * g);
  }
}

// ---------------------------------------------------------------------------
// Flash attention (causal, GQA). One wave per 16-query tile. Key blocks of
// 32 with online softmax; S = Q*K^T and O += P*V both on WMMA. K tiles are
// staged with async global->LDS DMA; V is transposed through VGPRs into the
// B-fragment-friendly [d][key] layout.
// q/k/v layout: (B, T, heads, 128) f16; output same layout (=> (B,T,2048)).
// ---------------------------------------------------------------------------
__global__ __launch_bounds__(32) void flash_attn_kernel(const half_t* __restrict__ qh,
                                                        const half_t* __restrict__ kh,
                                                        const half_t* __restrict__ vh,
                                                        half_t* __restrict__ oh,
                                                        int T) {
  __shared__ __align__(16) half_t Kt[32 * HEAD_DIM];   // [key][d]
  __shared__ __align__(16) half_t Vt[HEAD_DIM * 32];   // [d][key]  (transposed)
  __shared__ __align__(16) half_t Pt[16 * 32];         // [m][key]

  const int lane = threadIdx.x & 31;
  const int qt = blockIdx.x, h = blockIdx.y, b = blockIdx.z;
  const int kvh = h / (N_HEADS / N_KV_HEADS);
  const int q0 = qt * 16;
  const float scale = 0.08838834764831845f;  // 1/sqrt(128)

  // Q fragments (A layout), one per 32-wide K step of head_dim.
  v16h aq[4];
  {
    const int r  = q0 + (lane & 15);
    const int kb = (lane >> 4) << 3;
    const half_t* qrow = qh + (((size_t)(b * T + r)) * N_HEADS + h) * HEAD_DIM;
#pragma unroll
    for (int kd = 0; kd < 4; ++kd)
#pragma unroll
      for (int i = 0; i < 8; ++i) {
        aq[kd][i]     = qrow[kd * 32 + kb + i];
        aq[kd][8 + i] = qrow[kd * 32 + 16 + kb + i];
      }
  }

  float m_i[8], l_i[8];
  v8f zero = {};
  v8f acc[8];
#pragma unroll
  for (int v = 0; v < 8; ++v) { m_i[v] = -__builtin_inff(); l_i[v] = 0.0f; acc[v] = zero; }

  const int s_end = q0 + 16;  // causal: keys 0 .. q0+15
  for (int s0 = 0; s0 < s_end; s0 += 32) {
    // Stage K rows (async DMA, row-major) and V rows (transposed) into LDS.
    {
      const int key = lane;
      const half_t* ksrc = kh + (((size_t)(b * T + s0 + key)) * N_KV_HEADS + kvh) * HEAD_DIM;
      const half_t* vsrc = vh + (((size_t)(b * T + s0 + key)) * N_KV_HEADS + kvh) * HEAD_DIM;
#pragma unroll
      for (int i = 0; i < 16; ++i)
        async_copy_b128(Kt + key * HEAD_DIM + i * 8, ksrc + i * 8);
      const h4* vg = (const h4*)vsrc;
#pragma unroll
      for (int d4 = 0; d4 < 32; ++d4) {
        h4 vq = vg[d4];
        Vt[(d4 * 4 + 0) * 32 + key] = vq.x;
        Vt[(d4 * 4 + 1) * 32 + key] = vq.y;
        Vt[(d4 * 4 + 2) * 32 + key] = vq.z;
        Vt[(d4 * 4 + 3) * 32 + key] = vq.w;
      }
    }
    async_wait<0>();
    __syncthreads();

    // S tiles: 16 queries x 32 keys (two 16x16 WMMA outputs).
    v8f st[2];
#pragma unroll
    for (int j = 0; j < 2; ++j) {
      v8f c = zero;
#pragma unroll
      for (int kd = 0; kd < 4; ++kd)
        c = wmma_f16(aq[kd], frag_b(Kt, HEAD_DIM, j * 16, kd * 32), c);
      st[j] = c;
    }

    // Online softmax per row (row = v + 8*(lane>=16); cols across 16-lane group).
    const int c0 = s0 + (lane & 15);
    const int c1 = c0 + 16;
#pragma unroll
    for (int v = 0; v < 8; ++v) {
      const int m = q0 + v + ((lane >> 4) << 3);
      float s0v = (c0 <= m) ? st[0][v] * scale : -__builtin_inff();
      float s1v = (c1 <= m) ? st[1][v] * scale : -__builtin_inff();
      float rmax = fmaxf(s0v, s1v);
#pragma unroll
      for (int off = 8; off >= 1; off >>= 1) rmax = fmaxf(rmax, __shfl_xor(rmax, off, 32));
      const float mnew = fmaxf(m_i[v], rmax);
      const float a = __expf(m_i[v] - mnew);
      const float p0 = (c0 <= m) ? __expf(s0v - mnew) : 0.0f;
      const float p1 = (c1 <= m) ? __expf(s1v - mnew) : 0.0f;
      float rsum = p0 + p1;
#pragma unroll
      for (int off = 8; off >= 1; off >>= 1) rsum += __shfl_xor(rsum, off, 32);
      l_i[v] = l_i[v] * a + rsum;
      m_i[v] = mnew;
      const int ml = v + ((lane >> 4) << 3);
      Pt[ml * 32 + (lane & 15)]      = (half_t)p0;
      Pt[ml * 32 + 16 + (lane & 15)] = (half_t)p1;
#pragma unroll
      for (int n = 0; n < 8; ++n) acc[n][v] *= a;
    }
    __syncthreads();

    // O += P (16x32) * V (32x128): 8 N-subtiles.
    const v16h ap = frag_a(Pt, 32, 0, 0);
#pragma unroll
    for (int nd = 0; nd < 8; ++nd)
      acc[nd] = wmma_f16(ap, frag_b(Vt, 32, nd * 16, 0), acc[nd]);
    __syncthreads();
  }

  // Epilogue: O / l, store f16 as (B,T,H,128).
#pragma unroll
  for (int v = 0; v < 8; ++v) {
    const float inv_l = 1.0f / l_i[v];
    const int m = q0 + v + ((lane >> 4) << 3);
    half_t* orow = oh + (((size_t)(b * T + m)) * N_HEADS + h) * HEAD_DIM;
#pragma unroll
    for (int nd = 0; nd < 8; ++nd)
      orow[nd * 16 + (lane & 15)] = (half_t)(acc[nd][v] * inv_l);
  }
}

// ---------------------------------------------------------------------------
// Host orchestration.
// ---------------------------------------------------------------------------
extern "C" void kernel_launch(void* const* d_in, const int* in_sizes, int n_in,
                              void* d_out, int out_size, void* d_ws, size_t ws_size,
                              hipStream_t stream) {
  (void)in_sizes; (void)n_in; (void)out_size; (void)ws_size;

  const float* x    = (const float*)d_in[0];   // (4,2048,2048)
  const float* q_w  = (const float*)d_in[1];   // (2048,2048)
  const float* k_w  = (const float*)d_in[2];   // (512,2048)
  const float* v_w  = (const float*)d_in[3];   // (512,2048)
  const float* o_w  = (const float*)d_in[4];   // (2048,2048)
  const float* gain = (const float*)d_in[5];   // (16,)
  float* out = (float*)d_out;                  // (4,2048,2048)

  // Workspace carve-up (256B aligned).
  char* w = (char*)d_ws;
  size_t off = 0;
  auto take = [&](size_t bytes) -> void* {
    void* p = w + off;
    off += (bytes + 255) & ~(size_t)255;
    return p;
  };
  const size_t nX  = (size_t)BT_SZ * D_SZ;   // 16777216
  const size_t nQW = (size_t)D_SZ * D_SZ;    // 4194304
  const size_t nKW = (size_t)KV_D * D_SZ;    // 1048576
  half_t* xh  = (half_t*)take(nX  * 2);
  half_t* wqh = (half_t*)take(nQW * 2);
  half_t* wkh = (half_t*)take(nKW * 2);
  half_t* wvh = (half_t*)take(nKW * 2);
  half_t* woh = (half_t*)take(nQW * 2);
  float*  qf  = (float*) take(nX * 4);                       // q pre-norm f32
  float*  kf  = (float*) take((size_t)BT_SZ * KV_D * 4);     // k pre-norm f32
  half_t* vh  = (half_t*)take((size_t)BT_SZ * KV_D * 2);     // v f16 (direct)
  half_t* qh  = (half_t*)take(nX * 2);                       // q post rope f16
  half_t* kh  = (half_t*)take((size_t)BT_SZ * KV_D * 2);     // k post rope f16
  half_t* ah  = (half_t*)qf;  // reuse q-f32 region for attention output (f16)

  // 1) Convert inputs to f16.
  f32_to_f16_kernel<<<(int)(nX  / 4 / 256), 256, 0, stream>>>((const float4*)x,   (h4*)xh,  nX  / 4);
  f32_to_f16_kernel<<<(int)(nQW / 4 / 256), 256, 0, stream>>>((const float4*)q_w, (h4*)wqh, nQW / 4);
  f32_to_f16_kernel<<<(int)(nKW / 4 / 256), 256, 0, stream>>>((const float4*)k_w, (h4*)wkh, nKW / 4);
  f32_to_f16_kernel<<<(int)(nKW / 4 / 256), 256, 0, stream>>>((const float4*)v_w, (h4*)wvh, nKW / 4);
  f32_to_f16_kernel<<<(int)(nQW / 4 / 256), 256, 0, stream>>>((const float4*)o_w, (h4*)woh, nQW / 4);

  // 2) Projections (C = A * W^T). M=8192.
  gemm_nt_kernel<float ><<<dim3(D_SZ / 128, BT_SZ / 128), 256, 0, stream>>>(xh, wqh, qf, BT_SZ, D_SZ, D_SZ);
  gemm_nt_kernel<float ><<<dim3(KV_D / 128, BT_SZ / 128), 256, 0, stream>>>(xh, wkh, kf, BT_SZ, KV_D, D_SZ);
  gemm_nt_kernel<half_t><<<dim3(KV_D / 128, BT_SZ / 128), 256, 0, stream>>>(xh, wvh, vh, BT_SZ, KV_D, D_SZ);

  // 3) RMSNorm + RoPE (+gain for q). base rescaled since T=2048 > 1024:
  //    base = 10000 * 2^(128/126)  =>  log2(base) = log2(10000) + 128/126.
  const float log2_base = log2f(10000.0f) + 128.0f / 126.0f;
  rmsnorm_rope_kernel<<<(BT_SZ * N_HEADS) / 8, 256, 0, stream>>>(qf, qh, gain, N_HEADS, T_SZ, log2_base);
  rmsnorm_rope_kernel<<<(BT_SZ * N_KV_HEADS) / 8, 256, 0, stream>>>(kf, kh, nullptr, N_KV_HEADS, T_SZ, log2_base);

  // 4) Causal GQA flash attention.
  flash_attn_kernel<<<dim3(T_SZ / 16, N_HEADS, B_SZ), 32, 0, stream>>>(qh, kh, vh, ah, T_SZ);

  // 5) Output projection -> f32 result.
  gemm_nt_kernel<float><<<dim3(D_SZ / 128, BT_SZ / 128), 256, 0, stream>>>(ah, woh, out, BT_SZ, D_SZ, D_SZ);
}
